// ScaledAttention_91250875171484
// MI455X (gfx1250) — matrix operand order repaired and checked by reading, not compiled, not run on hardware
//
#include <hip/hip_runtime.h>
#include <hip/hip_bf16.h>

typedef __attribute__((ext_vector_type(8)))  __bf16 v8bf;
typedef __attribute__((ext_vector_type(16))) __bf16 v16bf;
typedef __attribute__((ext_vector_type(8)))  float  v8f;

#define LN_EPS 1e-5f

__device__ __forceinline__ v8f wmma_bf16(v16bf a, v16bf b, v8f c) {
  // (neg_a, A, neg_b, B, c_mod, C, reuse_a, reuse_b)
  return __builtin_amdgcn_wmma_f32_16x16x32_bf16(false, a, false, b, (short)0, c,
                                                 false, false);
}

// Build an A/B fragment from two 16-byte chunks (callers pass adjacent chunks: contiguous k = 16h + s).
__device__ __forceinline__ v16bf ld16_split(const __bf16* p0, const __bf16* p1) {
  v8bf lo = *(const v8bf*)p0;
  v8bf hi = *(const v8bf*)p1;
  v16bf r;
#pragma unroll
  for (int i = 0; i < 8; ++i) { r[i] = lo[i]; r[i + 8] = hi[i]; }
  return r;
}

// ---------------------------------------------------------------- precision prep
__global__ void f32_to_bf16_kernel(const float* __restrict__ in,
                                   __bf16* __restrict__ out, int n) {
  int i = blockIdx.x * blockDim.x + threadIdx.x;
  if (i < n) out[i] = (__bf16)in[i];
}

// out[c][r] = in[r][c]; out has shape [cols_in][rows_in]
__global__ void transpose_bf16_kernel(const float* __restrict__ in,
                                      __bf16* __restrict__ out,
                                      int rows_in, int cols_in) {
  int i = blockIdx.x * blockDim.x + threadIdx.x;
  int total = rows_in * cols_in;
  if (i >= total) return;
  int c = i / rows_in;
  int r = i - c * rows_in;
  out[i] = (__bf16)in[(size_t)r * cols_in + c];
}

// ---------------------------------------------------------------- QKV GEMM + LN
// xb:  [4096][1024] bf16 (rows = B*N)
// wt:  [3072][1024] bf16 (W_qkv transposed)
// Each wave: 16 rows x 64 cols = one (which,h) group for 16 tokens.
// which==0 -> LN + 0.125 scale -> qb[B*H][2048][64]
// which==1 -> LN               -> kb[B*H][2048][64]
// which==2 ->                  -> vt[B*H][64][2048]   (transposed)
__global__ __launch_bounds__(128)
void qkv_gemm_ln_kernel(const __bf16* __restrict__ xb, const __bf16* __restrict__ wt,
                        const float* __restrict__ qn_w, const float* __restrict__ qn_b,
                        const float* __restrict__ kn_w, const float* __restrict__ kn_b,
                        __bf16* __restrict__ qb, __bf16* __restrict__ kb,
                        __bf16* __restrict__ vt) {
  const int lane   = threadIdx.x & 31;
  const int wave   = threadIdx.x >> 5;
  const int lane15 = lane & 15;
  const int hi     = lane >> 4;
  const int r0     = blockIdx.x * 64 + wave * 16;   // global row tile (0..4095)
  const int g      = blockIdx.y;                    // 0..47
  const int which  = g >> 4;
  const int h      = g & 15;
  const int col0   = g * 64;

  v8f zero = {0.f, 0.f, 0.f, 0.f, 0.f, 0.f, 0.f, 0.f};
  v8f acc[4];
#pragma unroll
  for (int t = 0; t < 4; ++t) acc[t] = zero;

  const __bf16* arow = xb + (size_t)(r0 + lane15) * 1024;
  for (int k0 = 0; k0 < 1024; k0 += 32) {
    v16bf a = ld16_split(arow + k0 + 16 * hi, arow + k0 + 16 * hi + 8);
#pragma unroll
    for (int t = 0; t < 4; ++t) {
      const __bf16* bp =
          wt + (size_t)(col0 + t * 16 + lane15) * 1024 + 16 * hi + k0;
      acc[t] = wmma_bf16(a, *(const v16bf*)bp, acc[t]);
    }
  }

  const int bb    = r0 >> 11;        // batch
  const int nbase = r0 & 2047;       // sequence position base
  const size_t bh = (size_t)bb * 16 + h;

  if (which == 2) {
    // V: store transposed [bh][d][n]; 8 consecutive n per lane -> one b128 store
#pragma unroll
    for (int t = 0; t < 4; ++t) {
      v8bf pk;
#pragma unroll
      for (int r = 0; r < 8; ++r) pk[r] = (__bf16)acc[t][r];
      const int d = t * 16 + lane15;
      *(v8bf*)(vt + (bh * 64 + d) * 2048 + nbase + 8 * hi) = pk;
    }
  } else {
    // LayerNorm over the 64-wide head dim held in this wave's accumulators
    float mean[8], rstd[8];
#pragma unroll
    for (int r = 0; r < 8; ++r) {
      float s  = acc[0][r] + acc[1][r] + acc[2][r] + acc[3][r];
      float q2 = acc[0][r] * acc[0][r] + acc[1][r] * acc[1][r] +
                 acc[2][r] * acc[2][r] + acc[3][r] * acc[3][r];
      s  += __shfl_xor(s, 1);  s  += __shfl_xor(s, 2);
      s  += __shfl_xor(s, 4);  s  += __shfl_xor(s, 8);
      q2 += __shfl_xor(q2, 1); q2 += __shfl_xor(q2, 2);
      q2 += __shfl_xor(q2, 4); q2 += __shfl_xor(q2, 8);
      float mu  = s * (1.0f / 64.0f);
      float var = q2 * (1.0f / 64.0f) - mu * mu;
      mean[r] = mu;
      rstd[r] = rsqrtf(var + LN_EPS);
    }
    const float* wv = (which == 0) ? qn_w : kn_w;
    const float* bv = (which == 0) ? qn_b : kn_b;
    const float  sc = (which == 0) ? 0.125f : 1.0f;   // hd^-0.5 fused into q
    __bf16* dst = ((which == 0) ? qb : kb) + (bh * 2048 + nbase) * 64;
#pragma unroll
    for (int t = 0; t < 4; ++t) {
      const int   d   = t * 16 + lane15;
      const float w   = wv[d] * sc;
      const float bia = bv[d] * sc;
#pragma unroll
      for (int r = 0; r < 8; ++r) {
        float y = (acc[t][r] - mean[r]) * rstd[r] * w + bia;
        dst[(size_t)(r + 8 * hi) * 64 + d] = (__bf16)y;
      }
    }
  }
}

// ---------------------------------------------------------------- flash attention
// qb,kb: [B*H][2048][64] bf16 ; vt: [B*H][64][2048] bf16 ; ob: [B*N][1024] bf16
// One wave per 16-row q tile; online softmax; P staged through LDS to convert
// the 16x16 C-layout S tile into the A-layout operand for the PV WMMA.
__global__ __launch_bounds__(256)
void flash_attn_kernel(const __bf16* __restrict__ qb, const __bf16* __restrict__ kb,
                       const __bf16* __restrict__ vt, __bf16* __restrict__ ob) {
  __shared__ __bf16 lds_p[8][16 * 32];   // per-wave 16x32 P staging (8 KB)

  const int lane   = threadIdx.x & 31;
  const int wave   = threadIdx.x >> 5;
  const int lane15 = lane & 15;
  const int hi     = lane >> 4;
  const int bh     = blockIdx.y;          // 0..31
  const int b      = bh >> 4, h = bh & 15;
  const int q0     = blockIdx.x * 128 + wave * 16;

  const __bf16* kbh = kb + (size_t)bh * 2048 * 64;
  const __bf16* vbh = vt + (size_t)bh * 64 * 2048;

  const __bf16* qrow = qb + ((size_t)bh * 2048 + q0 + lane15) * 64;
  v16bf aq0 = ld16_split(qrow + 16 * hi,      qrow + 8 + 16 * hi);
  v16bf aq1 = ld16_split(qrow + 32 + 16 * hi, qrow + 40 + 16 * hi);

  v8f zero = {0.f, 0.f, 0.f, 0.f, 0.f, 0.f, 0.f, 0.f};
  v8f O[4];
#pragma unroll
  for (int t = 0; t < 4; ++t) O[t] = zero;
  float mx[8], lsum[8];
#pragma unroll
  for (int r = 0; r < 8; ++r) { mx[r] = -1e30f; lsum[r] = 0.f; }

  __bf16* pst = &lds_p[wave][0];

  for (int kc = 0; kc < 64; ++kc) {       // 32 keys per iteration
#pragma unroll
    for (int j = 0; j < 2; ++j) {         // two 16-key S tiles
      const int kt = kc * 2 + j;
      const __bf16* krow = kbh + (size_t)(kt * 16 + lane15) * 64 + 16 * hi;
      v8f s = zero;
      s = wmma_bf16(aq0, *(const v16bf*)krow, s);
      s = wmma_bf16(aq1, *(const v16bf*)(krow + 32), s);
#pragma unroll
      for (int r = 0; r < 8; ++r) {
        float t0 = s[r];
        float tm = t0;
        tm = fmaxf(tm, __shfl_xor(tm, 1));
        tm = fmaxf(tm, __shfl_xor(tm, 2));
        tm = fmaxf(tm, __shfl_xor(tm, 4));
        tm = fmaxf(tm, __shfl_xor(tm, 8));
        float mnew  = fmaxf(mx[r], tm);
        float alpha = __expf(mx[r] - mnew);
        mx[r] = mnew;
        float p  = __expf(t0 - mnew);
        float ps = p;
        ps += __shfl_xor(ps, 1); ps += __shfl_xor(ps, 2);
        ps += __shfl_xor(ps, 4); ps += __shfl_xor(ps, 8);
        lsum[r] = lsum[r] * alpha + ps;
        O[0][r] *= alpha; O[1][r] *= alpha; O[2][r] *= alpha; O[3][r] *= alpha;
        pst[(r + 8 * hi) * 32 + j * 16 + lane15] = (__bf16)p;
      }
    }
    // same-wave DS ops are in-order; wait for P stores then read in A layout
    asm volatile("s_wait_dscnt 0" ::: "memory");
    const __bf16* lp = pst + lane15 * 32 + 16 * hi;
    v16bf ap = ld16_split(lp, lp + 8);
    const __bf16* vrow = vbh + (size_t)lane15 * 2048 + kc * 32 + 16 * hi;
    O[0] = wmma_bf16(ap, *(const v16bf*)vrow,               O[0]);
    O[1] = wmma_bf16(ap, *(const v16bf*)(vrow + 16 * 2048), O[1]);
    O[2] = wmma_bf16(ap, *(const v16bf*)(vrow + 32 * 2048), O[2]);
    O[3] = wmma_bf16(ap, *(const v16bf*)(vrow + 48 * 2048), O[3]);
    asm volatile("" ::: "memory");        // keep next iter's P stores below the read
  }

  const int h64 = h * 64;
#pragma unroll
  for (int r = 0; r < 8; ++r) {
    const float inv = 1.0f / lsum[r];
    const int   n   = q0 + r + 8 * hi;
    __bf16* dst = ob + ((size_t)(b * 2048 + n)) * 1024 + h64 + lane15;
    dst[0]  = (__bf16)(O[0][r] * inv);
    dst[16] = (__bf16)(O[1][r] * inv);
    dst[32] = (__bf16)(O[2][r] * inv);
    dst[48] = (__bf16)(O[3][r] * inv);
  }
}

// ---------------------------------------------------------------- output projection
// ob:[4096][1024] bf16, wt:[1024][1024] bf16 (W_proj^T) -> out fp32 + bias
__global__ __launch_bounds__(128)
void proj_gemm_kernel(const __bf16* __restrict__ ob, const __bf16* __restrict__ wt,
                      const float* __restrict__ bias, float* __restrict__ out) {
  const int lane   = threadIdx.x & 31;
  const int wave   = threadIdx.x >> 5;
  const int lane15 = lane & 15;
  const int hi     = lane >> 4;
  const int r0     = blockIdx.x * 64 + wave * 16;
  const int col0   = blockIdx.y * 64;

  v8f zero = {0.f, 0.f, 0.f, 0.f, 0.f, 0.f, 0.f, 0.f};
  v8f acc[4];
#pragma unroll
  for (int t = 0; t < 4; ++t) acc[t] = zero;

  const __bf16* arow = ob + (size_t)(r0 + lane15) * 1024;
  for (int k0 = 0; k0 < 1024; k0 += 32) {
    v16bf a = ld16_split(arow + k0 + 16 * hi, arow + k0 + 16 * hi + 8);
#pragma unroll
    for (int t = 0; t < 4; ++t) {
      const __bf16* bp =
          wt + (size_t)(col0 + t * 16 + lane15) * 1024 + 16 * hi + k0;
      acc[t] = wmma_bf16(a, *(const v16bf*)bp, acc[t]);
    }
  }

#pragma unroll
  for (int t = 0; t < 4; ++t) {
    const int   c  = col0 + t * 16 + lane15;
    const float bi = bias[c];
#pragma unroll
    for (int r = 0; r < 8; ++r)
      out[(size_t)(r0 + r + 8 * hi) * 1024 + c] = acc[t][r] + bi;
  }
}

// ---------------------------------------------------------------- launch
extern "C" void kernel_launch(void* const* d_in, const int* in_sizes, int n_in,
                              void* d_out, int out_size, void* d_ws, size_t ws_size,
                              hipStream_t stream) {
  (void)in_sizes; (void)n_in; (void)out_size; (void)ws_size;
  const float* x      = (const float*)d_in[0];
  const float* Wqkv   = (const float*)d_in[1];
  const float* qn_w   = (const float*)d_in[2];
  const float* qn_b   = (const float*)d_in[3];
  const float* kn_w   = (const float*)d_in[4];
  const float* kn_b   = (const float*)d_in[5];
  const float* Wproj  = (const float*)d_in[6];
  const float* b_proj = (const float*)d_in[7];
  float* out = (float*)d_out;

  char* ws = (char*)d_ws;                           // ~36 MB used
  __bf16* xb     = (__bf16*)(ws);                   //  8 MB: x bf16 [4096][1024]
  __bf16* wqkvt  = (__bf16*)(ws + (8u  << 20));     //  6 MB: W_qkv^T
  __bf16* wprojt = (__bf16*)(ws + (14u << 20));     //  2 MB: W_proj^T
  __bf16* qb     = (__bf16*)(ws + (16u << 20));     //  4 MB: q (LN+scaled)
  __bf16* kb     = (__bf16*)(ws + (20u << 20));     //  4 MB: k (LN)
  __bf16* vt     = (__bf16*)(ws + (24u << 20));     //  4 MB: v transposed
  __bf16* obuf   = (__bf16*)(ws + (28u << 20));     //  8 MB: attention output

  f32_to_bf16_kernel<<<16384, 256, 0, stream>>>(x, xb, 4096 * 1024);
  transpose_bf16_kernel<<<12288, 256, 0, stream>>>(Wqkv, wqkvt, 1024, 3072);
  transpose_bf16_kernel<<<4096, 256, 0, stream>>>(Wproj, wprojt, 1024, 1024);
  qkv_gemm_ln_kernel<<<dim3(64, 48), 128, 0, stream>>>(
      xb, wqkvt, qn_w, qn_b, kn_w, kn_b, qb, kb, vt);
  flash_attn_kernel<<<dim3(16, 32), 256, 0, stream>>>(qb, kb, vt, obuf);
  proj_gemm_kernel<<<dim3(64, 16), 128, 0, stream>>>(obuf, wprojt, b_proj, out);
}
